// AMP_GCN_40664750358926
// MI455X (gfx1250) — matrix operand order repaired and checked by reading, not compiled, not run on hardware
//
#include <hip/hip_runtime.h>
#include <hip/hip_bf16.h>
#include <stdint.h>

#define N_NODES 100000
#define N_EDGES 1600000
#define N_GRAPHS 512
#define IN_DIM 320
#define HID 128
#define OUT_DIM 2

typedef __attribute__((ext_vector_type(16))) __bf16 v16bf;
typedef __attribute__((ext_vector_type(8)))  float  v8f;

union V16 { v16bf v; unsigned short u[16]; };

// gfx1250 has native bf16: let the backend pick v_cvt (packed where possible)
__device__ __forceinline__ __bf16 f2bf(float f) { return (__bf16)f; }

__device__ __forceinline__ unsigned short f2bf_bits(float f) {
  return __builtin_bit_cast(unsigned short, (__bf16)f);
}

// ---------------- degree / norm ----------------
__global__ void k_init_deg(float* __restrict__ deg) {
  int i = blockIdx.x * blockDim.x + threadIdx.x;
  if (i < N_NODES) deg[i] = 1.0f;            // self-loop
}

__global__ void k_edge_deg(const long long* __restrict__ ei, float* __restrict__ deg) {
  int e = blockIdx.x * blockDim.x + threadIdx.x;
  if (e < N_EDGES) atomicAdd(&deg[(int)ei[N_EDGES + e]], 1.0f);
}

__global__ void k_dinv(const float* __restrict__ deg, float* __restrict__ dinv) {
  int i = blockIdx.x * blockDim.x + threadIdx.x;
  if (i < N_NODES) {
    float d = deg[i];
    dinv[i] = (d > 0.0f) ? rsqrtf(d) : 0.0f;
  }
}

// ---------------- WMMA GEMM: Out[M,128] = A[M,K] @ W[K,128] ----------------
// 256 threads = 8 waves; wave w -> rows [blk*128 + w*16, +16), all 128 cols as
// eight 16x16 WMMA tiles. Entire W staged to LDS as bf16 ONCE per block
// (dynamic LDS: K*136*2 bytes; 87KB for K=320 -> 3 blocks/WGP), so the K loop
// runs barrier-free and A-loads pipeline under the WMMA burst.
#define SW_STRIDE 136   // pad 128->136 halves: lane stride 272B == 4 banks mod 64

__global__ __launch_bounds__(256) void k_gemm_wmma(
    const float* __restrict__ A, const float* __restrict__ W,
    float* __restrict__ Out, int M, int K) {
  extern __shared__ unsigned short sW[];     // [K][SW_STRIDE]

  const int lane = threadIdx.x & 31;
  const int wave = threadIdx.x >> 5;
  const int row0 = blockIdx.x * 128 + wave * 16;
  int m = row0 + (lane & 15);
  if (m > M - 1) m = M - 1;                  // clamp loads; stores predicated
  const int koff = (lane & 16) ? 8 : 0;

  // stage W[0:K, 0:128] -> LDS bf16 (convert once per block)
  for (int i = threadIdx.x; i < K * 128; i += 256) {
    int kk = i >> 7, nn = i & 127;
    sW[kk * SW_STRIDE + nn] = f2bf_bits(W[(size_t)i]);
  }
  __syncthreads();

  v8f zero = {};
  v8f acc[8];
#pragma unroll
  for (int t = 0; t < 8; ++t) acc[t] = zero;

  for (int kb = 0; kb < K; kb += 32) {
    // A fragment (16x32 bf16): lane holds row m, k = kb + {koff..+7, 16+koff..+7}
    const float* ap = A + (size_t)m * K + kb + koff;
    float4 a0 = *(const float4*)(ap);
    float4 a1 = *(const float4*)(ap + 4);
    float4 a2 = *(const float4*)(ap + 16);
    float4 a3 = *(const float4*)(ap + 20);
    v16bf av;
    av[0]  = f2bf(a0.x); av[1]  = f2bf(a0.y); av[2]  = f2bf(a0.z); av[3]  = f2bf(a0.w);
    av[4]  = f2bf(a1.x); av[5]  = f2bf(a1.y); av[6]  = f2bf(a1.z); av[7]  = f2bf(a1.w);
    av[8]  = f2bf(a2.x); av[9]  = f2bf(a2.y); av[10] = f2bf(a2.z); av[11] = f2bf(a2.w);
    av[12] = f2bf(a3.x); av[13] = f2bf(a3.y); av[14] = f2bf(a3.z); av[15] = f2bf(a3.w);

    // preload all 8 B fragments (distinct regs -> 128 strided 16-bit LDS loads in flight)
    const unsigned short* wcol = &sW[(kb + koff) * SW_STRIDE + (lane & 15)];  // B: lane holds column n, k = kb + {koff..+7, 16+koff..+7}
    V16 bf[8];
#pragma unroll
    for (int t = 0; t < 8; ++t) {
#pragma unroll
      for (int s = 0; s < 16; ++s)
        bf[t].u[s] = wcol[((s < 8) ? s : s + 8) * SW_STRIDE + t * 16];
    }
#pragma unroll
    for (int t = 0; t < 8; ++t) {
      acc[t] = __builtin_amdgcn_wmma_f32_16x16x32_bf16(
          false, av, false, bf[t].v, (short)0, acc[t], false, false);
    }
  }

  // D layout: VGPR r, lane -> m = row0 + r + 8*(lane/16), n = lane%16
  const int mbase = row0 + ((lane & 16) ? 8 : 0);
  const int n = lane & 15;
#pragma unroll
  for (int t = 0; t < 8; ++t) {
#pragma unroll
    for (int r = 0; r < 8; ++r) {
      int mo = mbase + r;
      if (mo < M) Out[(size_t)mo * HID + t * 16 + n] = acc[t][r];
    }
  }
}

// ---------------- aggregation ----------------
// out[i,:] = hw[i,:] * dinv[i]^2  (self loop term, also zero-initializes out)
__global__ void k_self_init(const float* __restrict__ hw, const float* __restrict__ dinv,
                            float* __restrict__ out) {
  size_t t = (size_t)blockIdx.x * blockDim.x + threadIdx.x;
  if (t < (size_t)N_NODES * HID) {
    int node = (int)(t >> 7);
    float w = dinv[node];
    out[t] = hw[t] * (w * w);
  }
}

// out[dst,:] += hw[src,:] * dinv[src]*dinv[dst]  -- one wave per edge, 4 feats/lane
__global__ void k_edge_agg(const long long* __restrict__ ei, const float* __restrict__ hw,
                           const float* __restrict__ dinv, float* __restrict__ out) {
  size_t tid = (size_t)blockIdx.x * blockDim.x + threadIdx.x;
  int e = (int)(tid >> 5);
  int lane = (int)(tid & 31);
  if (e < N_EDGES) {
    int s = (int)ei[e];
    int d = (int)ei[N_EDGES + e];
    float w = dinv[s] * dinv[d];
    float4 hv = *(const float4*)(hw + (size_t)s * HID + lane * 4);
    float* op = out + (size_t)d * HID + lane * 4;
    atomicAdd(op + 0, hv.x * w);
    atomicAdd(op + 1, hv.y * w);
    atomicAdd(op + 2, hv.z * w);
    atomicAdd(op + 3, hv.w * w);
  }
}

__global__ void k_bias_relu(float* __restrict__ h, const float* __restrict__ b) {
  size_t t = (size_t)blockIdx.x * blockDim.x + threadIdx.x;
  if (t < (size_t)N_NODES * HID) {
    int f = (int)(t & 127);
    h[t] = fmaxf(h[t] + b[f], 0.0f);
  }
}

// ---------------- pooling + FC ----------------
__global__ void k_zero(float* __restrict__ p, int n) {
  int t = blockIdx.x * blockDim.x + threadIdx.x;
  if (t < n) p[t] = 0.0f;
}

__global__ void k_pool_sum(const float* __restrict__ h, const long long* __restrict__ batch,
                           float* __restrict__ pooled) {
  size_t t = (size_t)blockIdx.x * blockDim.x + threadIdx.x;
  if (t < (size_t)N_NODES * HID) {
    int node = (int)(t >> 7);
    int f = (int)(t & 127);
    int g = (int)batch[node];
    atomicAdd(&pooled[g * HID + f], h[t]);
  }
}

__global__ void k_pool_cnt(const long long* __restrict__ batch, float* __restrict__ cnt) {
  int i = blockIdx.x * blockDim.x + threadIdx.x;
  if (i < N_NODES) atomicAdd(&cnt[(int)batch[i]], 1.0f);
}

__global__ void k_fc(const float* __restrict__ pooled, const float* __restrict__ cnt,
                     const float* __restrict__ Wfc, const float* __restrict__ bfc,
                     float* __restrict__ out) {
  int t = blockIdx.x * blockDim.x + threadIdx.x;
  if (t < N_GRAPHS * OUT_DIM) {
    int g = t >> 1, o = t & 1;
    float inv = 1.0f / fmaxf(cnt[g], 1.0f);
    float s = bfc[o];
    for (int k = 0; k < HID; ++k)
      s += pooled[g * HID + k] * inv * Wfc[k * OUT_DIM + o];
    out[g * OUT_DIM + o] = s;
  }
}

// ---------------- driver ----------------
extern "C" void kernel_launch(void* const* d_in, const int* in_sizes, int n_in,
                              void* d_out, int out_size, void* d_ws, size_t ws_size,
                              hipStream_t stream) {
  (void)in_sizes; (void)n_in; (void)out_size; (void)ws_size;
  const float*     x    = (const float*)d_in[0];
  const long long* ei   = (const long long*)d_in[1];
  const long long* bat  = (const long long*)d_in[2];
  const float*     W1   = (const float*)d_in[3];
  const float*     b1   = (const float*)d_in[4];
  const float*     W2   = (const float*)d_in[5];
  const float*     b2   = (const float*)d_in[6];
  const float*     Wfc  = (const float*)d_in[7];
  const float*     bfc  = (const float*)d_in[8];
  float* out = (float*)d_out;

  float* ws     = (float*)d_ws;
  float* deg    = ws;
  float* dinv   = ws + N_NODES;
  float* bufA   = ws + 2 * (size_t)N_NODES;                 // 16B-aligned: 800000 bytes in
  float* bufB   = bufA + (size_t)N_NODES * HID;
  float* pooled = bufB + (size_t)N_NODES * HID;
  float* cnt    = pooled + (size_t)N_GRAPHS * HID;

  const int TB = 256;
  const int gN   = (N_NODES + TB - 1) / TB;
  const int gE   = (N_EDGES + TB - 1) / TB;
  const int gNF  = (int)(((size_t)N_NODES * HID + TB - 1) / TB);
  const int gE32 = (int)(((size_t)N_EDGES * 32 + TB - 1) / TB);
  const int gGemm = (N_NODES + 127) / 128;
  const size_t lds1 = (size_t)IN_DIM * SW_STRIDE * sizeof(unsigned short); // 87040 B
  const size_t lds2 = (size_t)HID    * SW_STRIDE * sizeof(unsigned short); // 34816 B

  // degrees & normalization
  k_init_deg<<<gN, TB, 0, stream>>>(deg);
  k_edge_deg<<<gE, TB, 0, stream>>>(ei, deg);
  k_dinv<<<gN, TB, 0, stream>>>(deg, dinv);

  // layer 1: hw = x @ W1 ; aggregate ; +b1 ; relu
  k_gemm_wmma<<<gGemm, TB, lds1, stream>>>(x, W1, bufA, N_NODES, IN_DIM);
  k_self_init<<<gNF, TB, 0, stream>>>(bufA, dinv, bufB);
  k_edge_agg<<<gE32, TB, 0, stream>>>(ei, bufA, dinv, bufB);
  k_bias_relu<<<gNF, TB, 0, stream>>>(bufB, b1);

  // layer 2: hw = h1 @ W2 ; aggregate ; +b2 ; relu
  k_gemm_wmma<<<gGemm, TB, lds2, stream>>>(bufB, W2, bufA, N_NODES, HID);
  k_self_init<<<gNF, TB, 0, stream>>>(bufA, dinv, bufB);
  k_edge_agg<<<gE32, TB, 0, stream>>>(ei, bufA, dinv, bufB);
  k_bias_relu<<<gNF, TB, 0, stream>>>(bufB, b2);

  // mean-pool + FC head
  const int nPool = N_GRAPHS * HID + N_GRAPHS;
  k_zero<<<(nPool + TB - 1) / TB, TB, 0, stream>>>(pooled, nPool);  // pooled then cnt (contiguous)
  k_pool_sum<<<gNF, TB, 0, stream>>>(bufB, bat, pooled);
  k_pool_cnt<<<gN, TB, 0, stream>>>(bat, cnt);
  k_fc<<<(N_GRAPHS * OUT_DIM + TB - 1) / TB, TB, 0, stream>>>(pooled, cnt, Wfc, bfc, out);
}
